// Decoder_Layer_53738630807778
// MI455X (gfx1250) — compile-verified, matrix operand order checked by reading, not run on hardware
//
#include <hip/hip_runtime.h>
#include <cstdint>
#include <cstddef>

// ---------------------------------------------------------------------------
// Types
// ---------------------------------------------------------------------------
typedef __bf16 bf16;
typedef __bf16 v8bf  __attribute__((ext_vector_type(8)));
typedef __bf16 v16bf __attribute__((ext_vector_type(16)));
typedef float  v8f   __attribute__((ext_vector_type(8)));

// Problem constants (match reference)
#define DMODEL 1024
#define DFF    4096
#define NHEAD  16
#define HD     64
#define SEQL   6      // S+1
#define BATCH  2048
#define NSEQ   (6 * BATCH)          // 12288 sequences
#define ROWS_ALL (NSEQ * SEQL)      // 73728 token rows
#define SEL_ROWS 20480              // 2048*5 (noun_msg[0]) + 5*2048 (verb_msg[1:])

// ---------------------------------------------------------------------------
// fp32 -> bf16 round-to-nearest-even
// ---------------------------------------------------------------------------
__device__ __forceinline__ unsigned short f2bfbits(float f) {
  unsigned u = __builtin_bit_cast(unsigned, f);
  u += 0x7fffu + ((u >> 16) & 1u);
  return (unsigned short)(u >> 16);
}
__device__ __forceinline__ bf16 f2bf(float f) {
  unsigned short h = f2bfbits(f);
  return __builtin_bit_cast(bf16, h);
}

// ---------------------------------------------------------------------------
// Kernel 0: fp32 -> bf16 bulk convert (n4 = element_count / 4)
// ---------------------------------------------------------------------------
__global__ __launch_bounds__(256) void cvt_bf16(
    const float* __restrict__ in, bf16* __restrict__ outp, int n4)
{
  int i = blockIdx.x * 256 + threadIdx.x;
  if (i >= n4) return;
  float4 v = *(const float4*)(in + (size_t)i * 4);
  *(ushort4*)(outp + (size_t)i * 4) =
      make_ushort4(f2bfbits(v.x), f2bfbits(v.y), f2bfbits(v.z), f2bfbits(v.w));
}

// ---------------------------------------------------------------------------
// Kernel 1: build src = concat(verb, noun + role) -> bf16   (73728 x 1024)
// ---------------------------------------------------------------------------
__global__ __launch_bounds__(256) void build_src(
    const float* __restrict__ feat, const float* __restrict__ role,
    bf16* __restrict__ src)
{
  size_t e   = (size_t)blockIdx.x * 256 + threadIdx.x;   // float4 index
  size_t row = e >> 8;                                   // 256 float4 per row
  int    c   = (int)(e & 255) << 2;                      // column (float)
  int    l   = (int)(row % SEQL);
  size_t sb  = row / SEQL;                               // s*2048 + b
  float4 v = *(const float4*)(feat + row * DMODEL + c);
  if (l > 0) {
    float4 r = *(const float4*)(role + (sb * 5 + (size_t)(l - 1)) * DMODEL + c);
    v.x += r.x; v.y += r.y; v.z += r.z; v.w += r.w;
  }
  ushort4 pk = make_ushort4(f2bfbits(v.x), f2bfbits(v.y), f2bfbits(v.z), f2bfbits(v.w));
  *(ushort4*)(src + row * DMODEL + c) = pk;
}

// ---------------------------------------------------------------------------
// Kernel 2: pure-bf16 WMMA GEMM with ASYNC global->LDS staging:
//   C[M,N] = act(A[M,K] @ W[N,K]^T + bias)     A, W bf16; bias fp32
//   CBF: C stored as bf16 (else fp32);  ACT: 0 none, 1 relu, 2 sigmoid
// Tile 128x128x32, 256 threads (8 waves), wave tile 32x64 (2x4 wmma frags).
// Double-buffered LDS filled by GLOBAL_LOAD_ASYNC_TO_LDS_B128 (ASYNCcnt),
// so no staging VGPRs at all -> fits the 128-VGPR budget without spills.
// Requires M%128==0, N%128==0, K%32==0 (true for all calls here).
// ---------------------------------------------------------------------------
#define TM 128
#define TN 128
#define TK 32
#define LDSB 40   // bf16 elements per LDS row (32 + 8 pad): conflict-free b128

#define ASYNC_B128(ldsaddr, gaddr)                                             \
  asm volatile("global_load_async_to_lds_b128 %0, %1, off"                     \
               :: "v"(ldsaddr), "v"(gaddr) : "memory")
#define WAIT_ASYNC0() asm volatile("s_wait_asynccnt 0x0" ::: "memory")

template <int CBF, int ACT>
__global__ __launch_bounds__(256) void gemm_wmma(
    const bf16* __restrict__ A,
    const bf16* __restrict__ W,
    const float* __restrict__ bias,
    void* __restrict__ Cv,
    int M, int N, int K)
{
  __shared__ __align__(16) bf16 lA[2][TM * LDSB];
  __shared__ __align__(16) bf16 lB[2][TN * LDSB];

  const int tid  = threadIdx.x;
  const int lane = tid & 31;
  const int wave = tid >> 5;
  const int wm   = wave & 3;            // 4 waves along M (32 rows each)
  const int wn   = wave >> 2;           // 2 waves along N (64 cols each)
  const int l16  = lane & 15;
  const int kh   = (lane >> 4) << 3;    // K-half select for 16-bit A/B frags
  const int mhi  = (lane >> 4) << 3;    // C-frag row offset

  const size_t rowA0 = (size_t)blockIdx.y * TM;
  const size_t colB0 = (size_t)blockIdx.x * TN;

  // per-thread staging coords: 512 chunks of 8 bf16 per 128x32 tile
  const int sr = tid >> 2;              // row 0..63 / +64
  const int sc = (tid & 3) << 3;        // k-chunk 0,8,16,24

  // LDS destination byte addresses (low 32 bits of generic LDS address)
  const unsigned la0[2] = { (unsigned)(size_t)&lA[0][sr * LDSB + sc],
                            (unsigned)(size_t)&lA[1][sr * LDSB + sc] };
  const unsigned la1[2] = { (unsigned)(size_t)&lA[0][(sr + 64) * LDSB + sc],
                            (unsigned)(size_t)&lA[1][(sr + 64) * LDSB + sc] };
  const unsigned lb0[2] = { (unsigned)(size_t)&lB[0][sr * LDSB + sc],
                            (unsigned)(size_t)&lB[1][sr * LDSB + sc] };
  const unsigned lb1[2] = { (unsigned)(size_t)&lB[0][(sr + 64) * LDSB + sc],
                            (unsigned)(size_t)&lB[1][(sr + 64) * LDSB + sc] };

  auto stageAsync = [&](int k0, int buf) {
    const bf16* ga0 = A + (rowA0 + sr)      * (size_t)K + k0 + sc;
    const bf16* ga1 = A + (rowA0 + sr + 64) * (size_t)K + k0 + sc;
    const bf16* gb0 = W + (colB0 + sr)      * (size_t)K + k0 + sc;
    const bf16* gb1 = W + (colB0 + sr + 64) * (size_t)K + k0 + sc;
    ASYNC_B128(la0[buf], ga0);
    ASYNC_B128(la1[buf], ga1);
    ASYNC_B128(lb0[buf], gb0);
    ASYNC_B128(lb1[buf], gb1);
  };

  v8f acc[2][4];
  #pragma unroll
  for (int tm = 0; tm < 2; ++tm)
    #pragma unroll
    for (int tn = 0; tn < 4; ++tn) acc[tm][tn] = 0.0f;

  // prologue: stage tile 0
  stageAsync(0, 0);
  WAIT_ASYNC0();
  __syncthreads();

  int cur = 0;
  for (int k0 = 0; k0 < K; k0 += TK) {
    const bool hasNext = (k0 + TK) < K;
    if (hasNext) stageAsync(k0 + TK, cur ^ 1);   // lands under the WMMAs
    if ((k0 + 2 * TK) < K)
      __builtin_prefetch(W + (colB0 + (size_t)sr) * (size_t)K + k0 + 2 * TK, 0, 0);

    // ---- fragments (ds_load_b128 pairs) + 8 WMMA ----
    v16bf af[2], bfr[4];
    #pragma unroll
    for (int tm = 0; tm < 2; ++tm) {
      const bf16* p = &lA[cur][(wm * 32 + tm * 16 + l16) * LDSB + kh];
      v8bf lo = *(const v8bf*)p;
      v8bf hi = *(const v8bf*)(p + 16);
      af[tm] = __builtin_shufflevector(lo, hi, 0,1,2,3,4,5,6,7,8,9,10,11,12,13,14,15);
    }
    #pragma unroll
    for (int tn = 0; tn < 4; ++tn) {
      const bf16* p = &lB[cur][(wn * 64 + tn * 16 + l16) * LDSB + kh];
      v8bf lo = *(const v8bf*)p;
      v8bf hi = *(const v8bf*)(p + 16);
      bfr[tn] = __builtin_shufflevector(lo, hi, 0,1,2,3,4,5,6,7,8,9,10,11,12,13,14,15);
    }
    #pragma unroll
    for (int tm = 0; tm < 2; ++tm)
      #pragma unroll
      for (int tn = 0; tn < 4; ++tn)
        acc[tm][tn] = __builtin_amdgcn_wmma_f32_16x16x32_bf16(
            false, af[tm], false, bfr[tn], (short)0, acc[tm][tn], false, false);

    if (hasNext) {
      WAIT_ASYNC0();        // next tile fully landed in LDS
      __syncthreads();      // all waves done reading buf cur (and writing cur^1)
      cur ^= 1;
    }
  }

  // ---- epilogue: bias + activation + store ----
  #pragma unroll
  for (int tm = 0; tm < 2; ++tm) {
    #pragma unroll
    for (int tn = 0; tn < 4; ++tn) {
      size_t col = colB0 + wn * 64 + tn * 16 + l16;
      float  bc  = bias ? bias[col] : 0.0f;
      size_t mb  = rowA0 + wm * 32 + tm * 16 + mhi;
      #pragma unroll
      for (int i = 0; i < 8; ++i) {
        float v = acc[tm][tn][i] + bc;
        if (ACT == 1)      v = fmaxf(v, 0.0f);
        else if (ACT == 2) v = 1.0f / (1.0f + __expf(-v));
        size_t off = (mb + (size_t)i) * (size_t)N + col;
        if (CBF) ((bf16*)Cv)[off] = f2bf(v);
        else     ((float*)Cv)[off] = v;
      }
    }
  }
}

// ---------------------------------------------------------------------------
// Kernel 3: attention for the 20480 consumed rows only.
// One thread per (selected row, head). L=6, hd=64, scale = 1/8.
// Packed output rows: [0,10240)  = noun_msg[0]:  b*5 + (l-1)     (s=0)
//                     [10240, .) = verb_msg[1:]: 10240 + b*5 + (s-1) (l=0)
// ---------------------------------------------------------------------------
__global__ __launch_bounds__(256) void attn_small(
    const bf16* __restrict__ qkv, bf16* __restrict__ outp)
{
  int id = blockIdx.x * 256 + threadIdx.x;
  if (id >= SEL_ROWS * NHEAD) return;
  int head = id & (NHEAD - 1);
  int sel  = id >> 4;
  int s, b, l;
  if (sel < 10240) { s = 0; b = sel / 5; l = 1 + sel % 5; }
  else { int t = sel - 10240; b = t / 5; s = 1 + t % 5; l = 0; }

  size_t base = ((size_t)(s * BATCH + b)) * SEQL;        // first row of sequence
  const bf16* qp = qkv + (base + l) * (size_t)(3 * DMODEL) + head * HD;

  float q[HD];
  #pragma unroll
  for (int c = 0; c < 8; ++c) {
    v8bf v = *(const v8bf*)(qp + c * 8);
    #pragma unroll
    for (int i = 0; i < 8; ++i) q[c * 8 + i] = (float)v[i];
  }

  float sc[SEQL];
  #pragma unroll
  for (int lk = 0; lk < SEQL; ++lk) {
    const bf16* kp = qkv + (base + lk) * (size_t)(3 * DMODEL) + DMODEL + head * HD;
    float d = 0.0f;
    #pragma unroll
    for (int c = 0; c < 8; ++c) {
      v8bf v = *(const v8bf*)(kp + c * 8);
      #pragma unroll
      for (int i = 0; i < 8; ++i) d += q[c * 8 + i] * (float)v[i];
    }
    sc[lk] = d * 0.125f;
  }
  float mx = sc[0];
  #pragma unroll
  for (int lk = 1; lk < SEQL; ++lk) mx = fmaxf(mx, sc[lk]);
  float se = 0.0f;
  #pragma unroll
  for (int lk = 0; lk < SEQL; ++lk) { sc[lk] = __expf(sc[lk] - mx); se += sc[lk]; }
  float inv = 1.0f / se;

  float acc[HD];
  #pragma unroll
  for (int i = 0; i < HD; ++i) acc[i] = 0.0f;
  #pragma unroll
  for (int lk = 0; lk < SEQL; ++lk) {
    float w = sc[lk] * inv;
    const bf16* vp = qkv + (base + lk) * (size_t)(3 * DMODEL) + 2 * DMODEL + head * HD;
    #pragma unroll
    for (int c = 0; c < 8; ++c) {
      v8bf v = *(const v8bf*)(vp + c * 8);
      #pragma unroll
      for (int i = 0; i < 8; ++i) acc[c * 8 + i] += w * (float)v[i];
    }
  }
  bf16* op = outp + (size_t)sel * DMODEL + head * HD;
  #pragma unroll
  for (int c = 0; c < 16; ++c) {
    ushort4 pk = make_ushort4(f2bfbits(acc[c*4+0]), f2bfbits(acc[c*4+1]),
                              f2bfbits(acc[c*4+2]), f2bfbits(acc[c*4+3]));
    *(ushort4*)(op + c * 4) = pk;
  }
}

// ---------------------------------------------------------------------------
// LayerNorm helpers (one 256-thread block per row, D=1024, 4 elems/thread)
// ---------------------------------------------------------------------------
__device__ __forceinline__ float wave_sum(float v) {
  #pragma unroll
  for (int o = 16; o > 0; o >>= 1) v += __shfl_xor(v, o, 32);
  return v;
}

// h = LN(base_row + agg_b) ; writes fp32 and bf16 copies.
// base_row = base + (b*6 + rowOffset + j)*D, with r = b*rowsPerB + j
__global__ __launch_bounds__(256) void ln_pre(
    const float* __restrict__ base, const float* __restrict__ agg,
    const float* __restrict__ g, const float* __restrict__ be,
    float* __restrict__ h32, bf16* __restrict__ hbf,
    int rowsPerB, int rowOffset)
{
  int r = blockIdx.x;
  int b = r / rowsPerB, j = r % rowsPerB;
  const float* x = base + ((size_t)b * SEQL + rowOffset + j) * DMODEL;
  const float* a = agg + (size_t)b * DMODEL;
  int t = threadIdx.x << 2;
  float4 xv = *(const float4*)(x + t);
  float4 av = *(const float4*)(a + t);
  float v0 = xv.x + av.x, v1 = xv.y + av.y, v2 = xv.z + av.z, v3 = xv.w + av.w;
  float s  = v0 + v1 + v2 + v3;
  float sq = v0*v0 + v1*v1 + v2*v2 + v3*v3;

  __shared__ float red[16];
  float ws = wave_sum(s), wq = wave_sum(sq);
  int lane = threadIdx.x & 31, wave = threadIdx.x >> 5;
  if (lane == 0) { red[wave] = ws; red[8 + wave] = wq; }
  __syncthreads();
  float tot = 0.0f, totq = 0.0f;
  #pragma unroll
  for (int i = 0; i < 8; ++i) { tot += red[i]; totq += red[8 + i]; }
  float mu  = tot  * (1.0f / DMODEL);
  float var = totq * (1.0f / DMODEL) - mu * mu;
  float rs  = rsqrtf(var + 1e-5f);

  float4 gv = *(const float4*)(g + t);
  float4 bv = *(const float4*)(be + t);
  float y0 = (v0 - mu) * rs * gv.x + bv.x;
  float y1 = (v1 - mu) * rs * gv.y + bv.y;
  float y2 = (v2 - mu) * rs * gv.z + bv.z;
  float y3 = (v3 - mu) * rs * gv.w + bv.w;
  *(float4*)(h32 + (size_t)r * DMODEL + t) = make_float4(y0, y1, y2, y3);
  *(ushort4*)(hbf + (size_t)r * DMODEL + t) =
      make_ushort4(f2bfbits(y0), f2bfbits(y1), f2bfbits(y2), f2bfbits(y3));
}

// out_row = LN(h_row + ffn_row), scattered into d_out block 0.
__global__ __launch_bounds__(256) void ln_post(
    const float* __restrict__ h, const float* __restrict__ f,
    const float* __restrict__ g, const float* __restrict__ be,
    float* __restrict__ out0, int rowsPerB, int rowOffset)
{
  int r = blockIdx.x;
  int b = r / rowsPerB, j = r % rowsPerB;
  int t = threadIdx.x << 2;
  float4 xv = *(const float4*)(h + (size_t)r * DMODEL + t);
  float4 av = *(const float4*)(f + (size_t)r * DMODEL + t);
  float v0 = xv.x + av.x, v1 = xv.y + av.y, v2 = xv.z + av.z, v3 = xv.w + av.w;
  float s  = v0 + v1 + v2 + v3;
  float sq = v0*v0 + v1*v1 + v2*v2 + v3*v3;

  __shared__ float red[16];
  float ws = wave_sum(s), wq = wave_sum(sq);
  int lane = threadIdx.x & 31, wave = threadIdx.x >> 5;
  if (lane == 0) { red[wave] = ws; red[8 + wave] = wq; }
  __syncthreads();
  float tot = 0.0f, totq = 0.0f;
  #pragma unroll
  for (int i = 0; i < 8; ++i) { tot += red[i]; totq += red[8 + i]; }
  float mu  = tot  * (1.0f / DMODEL);
  float var = totq * (1.0f / DMODEL) - mu * mu;
  float rs  = rsqrtf(var + 1e-5f);

  float4 gv = *(const float4*)(g + t);
  float4 bv = *(const float4*)(be + t);
  float* dst = out0 + ((size_t)b * SEQL + rowOffset + j) * DMODEL + t;
  dst[0] = (v0 - mu) * rs * gv.x + bv.x;
  dst[1] = (v1 - mu) * rs * gv.y + bv.y;
  dst[2] = (v2 - mu) * rs * gv.z + bv.z;
  dst[3] = (v3 - mu) * rs * gv.w + bv.w;
}

// ---------------------------------------------------------------------------
// Host launcher
// ---------------------------------------------------------------------------
extern "C" void kernel_launch(void* const* d_in, const int* in_sizes, int n_in,
                              void* d_out, int out_size, void* d_ws, size_t ws_size,
                              hipStream_t stream) {
  const float* features = (const float*)d_in[0];
  const float* role     = (const float*)d_in[1];
  const float* w_in     = (const float*)d_in[2];
  const float* b_in     = (const float*)d_in[3];
  const float* w_out    = (const float*)d_in[4];
  const float* b_out    = (const float*)d_in[5];
  const float* ln1_g = (const float*)d_in[6],  *ln1_b = (const float*)d_in[7];
  const float* ln2_g = (const float*)d_in[8],  *ln2_b = (const float*)d_in[9];
  const float* ln3_g = (const float*)d_in[10], *ln3_b = (const float*)d_in[11];
  const float* ln4_g = (const float*)d_in[12], *ln4_b = (const float*)d_in[13];
  const float* ffn1_w1 = (const float*)d_in[14], *ffn1_b1 = (const float*)d_in[15];
  const float* ffn1_w2 = (const float*)d_in[16], *ffn1_b2 = (const float*)d_in[17];
  const float* ffn2_w1 = (const float*)d_in[18], *ffn2_b1 = (const float*)d_in[19];
  const float* ffn2_w2 = (const float*)d_in[20], *ffn2_b2 = (const float*)d_in[21];
  const float* agg1_w  = (const float*)d_in[22], *agg1_b  = (const float*)d_in[23];
  const float* agg2_w  = (const float*)d_in[24], *agg2_b  = (const float*)d_in[25];

  char* ws = (char*)d_ws;

  // --- persistent bf16 weight region (converted once per launch) ---
  bf16* wbf_in   = (bf16*)(ws + 0);            // 3072x1024
  bf16* wbf_out  = (bf16*)(ws + 6291456);      // 1024x1024
  bf16* wbf_ag1  = (bf16*)(ws + 8388608);      // 1024x5120
  bf16* wbf_ag2  = (bf16*)(ws + 18874368);     // 1024x5120
  bf16* wbf_f1a  = (bf16*)(ws + 29360128);     // 4096x1024
  bf16* wbf_f1b  = (bf16*)(ws + 37748736);     // 1024x4096
  bf16* wbf_f2a  = (bf16*)(ws + 46137344);     // 4096x1024
  bf16* wbf_f2b  = (bf16*)(ws + 54525952);     // 1024x4096
  // --- Region A (reused): src_bf16, then attention outputs ---
  const size_t A0 = 62914560;
  bf16* srcbf  = (bf16*)(ws + A0);             // 73728*1024 bf16
  bf16* obf    = (bf16*)(ws + A0);             // 20480*1024 bf16 (src dead)
  bf16* attnbf = (bf16*)(ws + A0 + 41943040);  // 20480*1024 bf16 (post out-proj)
  // --- Region B (reused): qkv_bf16, then post-attention buffers ---
  const size_t B0 = A0 + 150994944;
  bf16*  qkvbf = (bf16*)(ws + B0);             // 73728*3072 bf16
  float* aggv  = (float*)(ws + B0 + 0);
  float* nagg  = (float*)(ws + B0 + 8388608);
  float* h32   = (float*)(ws + B0 + 16777216); // 10240*1024 f32
  bf16*  hbf   = (bf16*) (ws + B0 + 58720256); // 10240*1024 bf16
  float* s32   = (float*)(ws + B0 + 79691776); // 2048*1024 f32
  bf16*  sbf   = (bf16*) (ws + B0 + 88080384); // 2048*1024 bf16
  bf16*  mid1  = (bf16*) (ws + B0 + 92274688); // 10240*4096 bf16
  float* f1o   = (float*)(ws + B0 + 176160768);// 10240*1024 f32
  bf16*  mid2  = (bf16*) (ws + B0 + 218103808);// 2048*4096 bf16
  float* f2o   = (float*)(ws + B0 + 234881024);// 2048*1024 f32

  float* out = (float*)d_out;

  // 0) pass-through copy; block 0 overwritten by ln_post at the end
  hipMemcpyAsync(d_out, (const void*)features, (size_t)out_size * sizeof(float),
                 hipMemcpyDeviceToDevice, stream);

  // 0b) convert all weights to bf16 once
  auto cvt = [&](const float* src, bf16* dst, int elems) {
    int n4 = elems / 4;
    cvt_bf16<<<(n4 + 255) / 256, 256, 0, stream>>>(src, dst, n4);
  };
  cvt(w_in,    wbf_in,  3 * DMODEL * DMODEL);
  cvt(w_out,   wbf_out, DMODEL * DMODEL);
  cvt(agg1_w,  wbf_ag1, DMODEL * 5 * DMODEL);
  cvt(agg2_w,  wbf_ag2, DMODEL * 5 * DMODEL);
  cvt(ffn1_w1, wbf_f1a, DFF * DMODEL);
  cvt(ffn1_w2, wbf_f1b, DMODEL * DFF);
  cvt(ffn2_w1, wbf_f2a, DFF * DMODEL);
  cvt(ffn2_w2, wbf_f2b, DMODEL * DFF);

  // 1) src = [verb, noun + role] in bf16
  build_src<<<ROWS_ALL, 256, 0, stream>>>(features, role, srcbf);

  // 2) QKV: (73728 x 3072) = src @ w_in^T + b_in  -> bf16
  gemm_wmma<1, 0><<<dim3(3 * DMODEL / TN, ROWS_ALL / TM), 256, 0, stream>>>(
      srcbf, wbf_in, b_in, qkvbf, ROWS_ALL, 3 * DMODEL, DMODEL);

  // 3) attention for the 20480 consumed rows -> packed bf16
  attn_small<<<(SEL_ROWS * NHEAD + 255) / 256, 256, 0, stream>>>(qkvbf, obf);

  // 4) out projection: (20480 x 1024) -> bf16 (feeds agg GEMMs directly)
  gemm_wmma<1, 0><<<dim3(DMODEL / TN, SEL_ROWS / TM), 256, 0, stream>>>(
      obf, wbf_out, b_out, attnbf, SEL_ROWS, DMODEL, DMODEL);

  // 5) aggregations: packed rows give contiguous (2048, 5120) views; sigmoid
  gemm_wmma<0, 2><<<dim3(DMODEL / TN, BATCH / TM), 256, 0, stream>>>(
      attnbf + (size_t)10240 * DMODEL, wbf_ag1, agg1_b, aggv,
      BATCH, DMODEL, 5 * DMODEL);
  gemm_wmma<0, 2><<<dim3(DMODEL / TN, BATCH / TM), 256, 0, stream>>>(
      attnbf, wbf_ag2, agg2_b, nagg, BATCH, DMODEL, 5 * DMODEL);

  // 6) h = LN1(tgt_noun + agg_verb) ; s = LN3(tgt_verb + noun_agg)
  ln_pre<<<BATCH * 5, 256, 0, stream>>>(features, aggv, ln1_g, ln1_b, h32, hbf, 5, 1);
  ln_pre<<<BATCH,     256, 0, stream>>>(features, nagg, ln3_g, ln3_b, s32, sbf, 1, 0);

  // 7) FFN1 (noun): relu GEMM then down projection
  gemm_wmma<1, 1><<<dim3(DFF / TN, BATCH * 5 / TM), 256, 0, stream>>>(
      hbf, wbf_f1a, ffn1_b1, mid1, BATCH * 5, DFF, DMODEL);
  gemm_wmma<0, 0><<<dim3(DMODEL / TN, BATCH * 5 / TM), 256, 0, stream>>>(
      mid1, wbf_f1b, ffn1_b2, f1o, BATCH * 5, DMODEL, DFF);

  // 8) FFN2 (verb)
  gemm_wmma<1, 1><<<dim3(DFF / TN, BATCH / TM), 256, 0, stream>>>(
      sbf, wbf_f2a, ffn2_b1, mid2, BATCH, DFF, DMODEL);
  gemm_wmma<0, 0><<<dim3(DMODEL / TN, BATCH / TM), 256, 0, stream>>>(
      mid2, wbf_f2b, ffn2_b2, f2o, BATCH, DMODEL, DFF);

  // 9) final LNs, scattered into d_out block 0
  ln_post<<<BATCH * 5, 256, 0, stream>>>(h32, f1o, ln2_g, ln2_b, out, 5, 1);
  ln_post<<<BATCH,     256, 0, stream>>>(s32, f2o, ln4_g, ln4_b, out, 1, 0);

  (void)n_in; (void)in_sizes; (void)ws_size;
}
